// CTM_partpad_dict_BN_6030134083848
// MI455X (gfx1250) — matrix operand order repaired and checked by reading, not compiled, not run on hardware
//
#include <hip/hip_runtime.h>

// ---------------------------------------------------------------------------
// CTM block (token2map -> 3x3 s2 conv -> map2token + skip -> BN -> conf ->
// grid clustering) for MI455X (gfx1250, wave32, WMMA).
// - GEMMs use v_wmma_f32_16x16x32_bf16 with LDS-staged A tiles shared by all
//   8 waves of a workgroup; conv uses 32-pixel M-tiles so B (weight) fragments
//   are reused across two M-subtiles (halves L2 weight traffic).
// - LDS A rows padded to 136 bf16 (68 dwords) to avoid bank aliasing.
// - Skip-GEMM A staging uses global_load_async_to_lds_b128 (ASYNCcnt) since
//   it needs no boundary predication; fenced with s_wait_asynccnt.
// - Scatters/gathers are wave-per-point with f32 global atomics.
// ---------------------------------------------------------------------------

typedef __attribute__((ext_vector_type(16))) __bf16 v16bf;
typedef __attribute__((ext_vector_type(8)))  float  v8f;

union Frag { v16bf v; uint4 q[2]; };

#define DEVI __device__ __forceinline__

constexpr int B_ = 4, N_ = 16384, CI = 128, CO = 256;
constexpr int H_ = 256, W_ = 256, N0 = 65536;
constexpr int HD = 128, WD = 128, HC = 64, WC = 64, NS = HC * WC;
constexpr int LDSPITCH = 136;  // bf16 elements per LDS A-row (128 + 8 pad)
constexpr float EPSF = 1e-6f, BNEPS = 1e-5f;

DEVI unsigned short f2bf(float f) {
  unsigned int u = __float_as_uint(f);
  return (unsigned short)((u + 0x7FFFu + ((u >> 16) & 1u)) >> 16);
}

DEVI int grid_cell(float lx, float ly, int Hg, int Wg) {
  lx = fminf(fmaxf(lx, -1.f), 1.f);
  ly = fminf(fmaxf(ly, -1.f), 1.f);
  int px = (int)rintf(0.5f * (lx + 1.f) * (float)Wg - 0.5f);
  int py = (int)rintf(0.5f * (ly + 1.f) * (float)Hg - 0.5f);
  px = min(max(px, 0), Wg - 1);
  py = min(max(py, 0), Hg - 1);
  return py * Wg + px;
}

DEVI v8f wmma_bf16(const Frag& a, const Frag& b, v8f acc) {
  return __builtin_amdgcn_wmma_f32_16x16x32_bf16(
      false, a.v, false, b.v, (short)0, acc, false, false);
}

// -------------------------- packing kernels --------------------------------

// conv_w [3][3][CI][CO] f32 -> wq [CO][9*CI] bf16 (K contiguous per column)
__global__ void k_pack_convw(const float* __restrict__ w, unsigned short* __restrict__ wq) {
  int i = blockIdx.x * 256 + threadIdx.x;
  if (i >= CO * 9 * CI) return;
  int n = i / (9 * CI);
  int k = i % (9 * CI);
  int tap = k / CI, ci = k % CI;
  wq[i] = f2bf(w[(tap * CI + ci) * CO + n]);
}

// skip_w [CO][CI] f32 -> bf16 (already [N][K])
__global__ void k_pack_skipw(const float* __restrict__ w, unsigned short* __restrict__ sw) {
  int i = blockIdx.x * 256 + threadIdx.x;
  if (i >= CO * CI) return;
  sw[i] = f2bf(w[i]);
}

// x [B][N][CI] f32 -> bf16
__global__ void k_pack_x(const float* __restrict__ x, unsigned short* __restrict__ xb) {
  int i = blockIdx.x * 256 + threadIdx.x;
  if (i >= B_ * N_ * CI) return;
  xb[i] = f2bf(x[i]);
}

// -------------------------- token2map --------------------------------------

// wave per point; lane handles 4 channels of CI=128
__global__ void k_token2map_scatter(const float* __restrict__ x,
                                    const float* __restrict__ loc,
                                    const int* __restrict__ idx_agg,
                                    float* __restrict__ ssum,
                                    float* __restrict__ cnt) {
  int pt = blockIdx.x * 8 + (threadIdx.x >> 5);
  int lane = threadIdx.x & 31;
  int b = pt / N0, n0 = pt % N0;
  float lx = loc[(size_t)(b * N0 + n0) * 2 + 0];
  float ly = loc[(size_t)(b * N0 + n0) * 2 + 1];
  int cell = grid_cell(lx, ly, H_, W_);
  int tok = idx_agg[b * N0 + n0];
  const float4* src = (const float4*)(x + (size_t)(b * N_ + tok) * CI);
  float4 f = src[lane];
  float* dst = ssum + (size_t)(b * (H_ * W_) + cell) * CI + lane * 4;
  atomicAdd(dst + 0, f.x);
  atomicAdd(dst + 1, f.y);
  atomicAdd(dst + 2, f.z);
  atomicAdd(dst + 3, f.w);
  if (lane == 0) atomicAdd(cnt + (size_t)b * (H_ * W_) + cell, 1.0f);
}

// normalize and convert map to bf16 NHWC
__global__ void k_map_norm(const float* __restrict__ ssum,
                           const float* __restrict__ cnt,
                           unsigned short* __restrict__ xmapb) {
  size_t i = (size_t)blockIdx.x * 256 + threadIdx.x;
  if (i >= (size_t)B_ * H_ * W_ * CI) return;
  size_t hw = i / CI;
  xmapb[i] = f2bf(ssum[i] / (cnt[hw] + EPSF));
}

// -------------------------- conv 3x3 s2 (implicit GEMM, WMMA bf16) ---------
// Block: 256 threads = 8 waves; block computes 32 output pixels x all 256
// output channels. A (32 rows x 9*128 K) staged once in LDS, shared by all
// waves; each wave: 2 M-subtiles x 2 N-tiles -> 144 WMMAs, B frags reused
// across the two M-subtiles.

__global__ void __launch_bounds__(256)
k_conv_wmma(const unsigned short* __restrict__ xmap,
            const unsigned short* __restrict__ wq,
            const float* __restrict__ conv_b,
            float* __restrict__ y) {
  __shared__ unsigned short a_lds[9 * 32 * LDSPITCH];

  int t = blockIdx.x;          // B * HD * (WD/32) = 2048 M-tiles
  int oxt = t & 3;             // 32-pixel tile along WD
  int oy = (t >> 2) & 127;
  int b = t >> 9;
  int tid = threadIdx.x;

  // ---- stage A into LDS: thread = (row, 16B chunk); 2 row-groups x 9 taps
  {
    int chunk = tid & 15;      // 16B chunk of the 256B channel row
#pragma unroll
    for (int rr = 0; rr < 2; ++rr) {
      int row = (tid >> 4) + rr * 16;   // 0..31 output pixel within tile
      int ox = oxt * 32 + row;
#pragma unroll
      for (int tap = 0; tap < 9; ++tap) {
        int ky = tap / 3, kx = tap % 3;
        int iy = 2 * oy - 1 + ky;
        int ix = 2 * ox - 1 + kx;
        uint4 v = make_uint4(0u, 0u, 0u, 0u);
        if (((unsigned)iy < (unsigned)H_) && ((unsigned)ix < (unsigned)W_)) {
          v = *(const uint4*)(xmap + ((size_t)(b * H_ + iy) * W_ + ix) * CI + chunk * 8);
        }
        *(uint4*)(a_lds + (tap * 32 + row) * LDSPITCH + chunk * 8) = v;
      }
    }
  }
  __syncthreads();

  int lane = tid & 31;
  int wv = tid >> 5;           // 0..7 -> N columns [wv*32, wv*32+32)
  int half = lane >> 4;
  int mrow = lane & 15;
  int n0 = wv * 32;

  v8f acc00 = {}, acc01 = {}, acc10 = {}, acc11 = {};
#pragma unroll
  for (int tap = 0; tap < 9; ++tap) {
    const unsigned short* aR0 = a_lds + (tap * 32 + mrow) * LDSPITCH;
    const unsigned short* aR1 = a_lds + (tap * 32 + 16 + mrow) * LDSPITCH;
    const unsigned short* b0 = wq + (size_t)(n0 + mrow) * (9 * CI) + tap * CI;
    const unsigned short* b1 = wq + (size_t)(n0 + 16 + mrow) * (9 * CI) + tap * CI;
#pragma unroll
    for (int c = 0; c < 4; ++c) {
      Frag a0, a1, f0, f1;
      a0.q[0] = *(const uint4*)(aR0 + c * 32 + half * 8);
      a0.q[1] = *(const uint4*)(aR0 + c * 32 + 16 + half * 8);
      a1.q[0] = *(const uint4*)(aR1 + c * 32 + half * 8);
      a1.q[1] = *(const uint4*)(aR1 + c * 32 + 16 + half * 8);
      f0.q[0] = *(const uint4*)(b0 + c * 32 + half * 8);
      f0.q[1] = *(const uint4*)(b0 + c * 32 + 16 + half * 8);
      f1.q[0] = *(const uint4*)(b1 + c * 32 + half * 8);
      f1.q[1] = *(const uint4*)(b1 + c * 32 + 16 + half * 8);
      acc00 = wmma_bf16(a0, f0, acc00);
      acc01 = wmma_bf16(a0, f1, acc01);
      acc10 = wmma_bf16(a1, f0, acc10);
      acc11 = wmma_bf16(a1, f1, acc11);
    }
  }

  int col0 = n0 + mrow;
  int col1 = n0 + 16 + mrow;
  float bias0 = conv_b[col0];
  float bias1 = conv_b[col1];
#pragma unroll
  for (int r = 0; r < 8; ++r) {
    int m = half * 8 + r;
    size_t base0 = ((size_t)(b * HD + oy) * WD + oxt * 32 + m) * CO;
    size_t base1 = ((size_t)(b * HD + oy) * WD + oxt * 32 + 16 + m) * CO;
    y[base0 + col0] = acc00[r] + bias0;
    y[base0 + col1] = acc01[r] + bias1;
    y[base1 + col0] = acc10[r] + bias0;
    y[base1 + col1] = acc11[r] + bias1;
  }
}

// -------------------------- map2token (bilinear gather + scatter) ----------

__global__ void k_map2token_scatter(const float* __restrict__ y,
                                    const float* __restrict__ loc,
                                    const int* __restrict__ idx_agg,
                                    const float* __restrict__ aggw,
                                    float* __restrict__ toknum,
                                    float* __restrict__ tokden) {
  int pt = blockIdx.x * 8 + (threadIdx.x >> 5);
  int lane = threadIdx.x & 31;
  int b = pt / N0, n0 = pt % N0;
  float lx = fminf(fmaxf(loc[(size_t)(b * N0 + n0) * 2 + 0], -1.f), 1.f);
  float ly = fminf(fmaxf(loc[(size_t)(b * N0 + n0) * 2 + 1], -1.f), 1.f);
  float fx = fminf(fmaxf(0.5f * (lx + 1.f) * (float)WD - 0.5f, 0.f), (float)(WD - 1));
  float fy = fminf(fmaxf(0.5f * (ly + 1.f) * (float)HD - 0.5f, 0.f), (float)(HD - 1));
  float x0f = floorf(fx), y0f = floorf(fy);
  float wx = fx - x0f, wy = fy - y0f;
  int x0 = (int)x0f, y0 = (int)y0f;
  int x1 = min(x0 + 1, WD - 1), y1 = min(y0 + 1, HD - 1);
  const float* base = y + (size_t)b * HD * WD * CO;
  int c0 = lane * 8;
  const float4* p00 = (const float4*)(base + ((size_t)y0 * WD + x0) * CO + c0);
  const float4* p10 = (const float4*)(base + ((size_t)y0 * WD + x1) * CO + c0);
  const float4* p01 = (const float4*)(base + ((size_t)y1 * WD + x0) * CO + c0);
  const float4* p11 = (const float4*)(base + ((size_t)y1 * WD + x1) * CO + c0);
  float w00 = (1.f - wx) * (1.f - wy), w10 = wx * (1.f - wy);
  float w01 = (1.f - wx) * wy, w11 = wx * wy;
  float wpt = aggw[b * N0 + n0];
  int tok = idx_agg[b * N0 + n0];
  float* dst = toknum + (size_t)(b * N_ + tok) * CO + c0;
#pragma unroll
  for (int h = 0; h < 2; ++h) {
    float4 a00 = p00[h], a10 = p10[h], a01 = p01[h], a11 = p11[h];
    float f0 = w00 * a00.x + w10 * a10.x + w01 * a01.x + w11 * a11.x;
    float f1 = w00 * a00.y + w10 * a10.y + w01 * a01.y + w11 * a11.y;
    float f2 = w00 * a00.z + w10 * a10.z + w01 * a01.z + w11 * a11.z;
    float f3 = w00 * a00.w + w10 * a10.w + w01 * a01.w + w11 * a11.w;
    atomicAdd(dst + h * 4 + 0, f0 * wpt);
    atomicAdd(dst + h * 4 + 1, f1 * wpt);
    atomicAdd(dst + h * 4 + 2, f2 * wpt);
    atomicAdd(dst + h * 4 + 3, f3 * wpt);
  }
  if (lane == 0) atomicAdd(tokden + (size_t)b * N_ + tok, wpt);
}

// -------------------------- skip GEMM (WMMA bf16) + combine ----------------
// Block: 256 threads = 8 waves; 16 token rows x all 256 output channels.
// A tile (16 x 128 bf16) staged in LDS via async global->LDS copy (all
// lanes in-bounds; 16B per lane), fenced with s_wait_asynccnt.

__global__ void __launch_bounds__(256)
k_skip_wmma(const unsigned short* __restrict__ xb,
            const unsigned short* __restrict__ sw,
            const float* __restrict__ tokden,
            float* __restrict__ tok) {  // in: num, out: map2token + skip
  __shared__ unsigned short a_lds[16 * LDSPITCH];

  int row0 = blockIdx.x * 16;
  int tid = threadIdx.x;
  {
    int row = tid >> 4;
    int chunk = tid & 15;
    unsigned ldsa = (unsigned)(size_t)(a_lds + row * LDSPITCH + chunk * 8);
    unsigned long long ga =
        (unsigned long long)(size_t)(xb + (size_t)(row0 + row) * CI + chunk * 8);
    asm volatile("global_load_async_to_lds_b128 %0, %1, off"
                 :: "v"(ldsa), "v"(ga)
                 : "memory");
    asm volatile("s_wait_asynccnt 0x0" ::: "memory");
  }
  __syncthreads();

  int lane = tid & 31;
  int wv = tid >> 5;
  int half = lane >> 4;
  int mrow = lane & 15;
  int n0 = wv * 32;

  const unsigned short* aRow = a_lds + mrow * LDSPITCH;
  const unsigned short* b0 = sw + (size_t)(n0 + mrow) * CI;
  const unsigned short* b1 = sw + (size_t)(n0 + 16 + mrow) * CI;

  v8f acc0 = {}, acc1 = {};
#pragma unroll
  for (int c = 0; c < 4; ++c) {
    Frag a, f0, f1;
    a.q[0] = *(const uint4*)(aRow + c * 32 + half * 8);
    a.q[1] = *(const uint4*)(aRow + c * 32 + 16 + half * 8);
    f0.q[0] = *(const uint4*)(b0 + c * 32 + half * 8);
    f0.q[1] = *(const uint4*)(b0 + c * 32 + 16 + half * 8);
    f1.q[0] = *(const uint4*)(b1 + c * 32 + half * 8);
    f1.q[1] = *(const uint4*)(b1 + c * 32 + 16 + half * 8);
    acc0 = wmma_bf16(a, f0, acc0);
    acc1 = wmma_bf16(a, f1, acc1);
  }

  int col0 = n0 + mrow;
  int col1 = n0 + 16 + mrow;
#pragma unroll
  for (int r = 0; r < 8; ++r) {
    int row = row0 + half * 8 + r;
    float rd = 1.0f / (tokden[row] + EPSF);
    size_t o0 = (size_t)row * CO + col0;
    size_t o1 = (size_t)row * CO + col1;
    tok[o0] = tok[o0] * rd + acc0[r];
    tok[o1] = tok[o1] * rd + acc1[r];
  }
}

// -------------------------- BatchNorm + conf -------------------------------

__global__ void k_bn_stats(const float* __restrict__ tok, float* __restrict__ stats) {
  int c = threadIdx.x;
  int r0 = blockIdx.x * 256;
  float s = 0.f, s2 = 0.f;
  for (int r = 0; r < 256; ++r) {
    float v = tok[(size_t)(r0 + r) * CO + c];
    s += v;
    s2 += v * v;
  }
  atomicAdd(stats + c, s);
  atomicAdd(stats + CO + c, s2);
}

__global__ void k_bn_conf(const float* __restrict__ stats,
                          const float* __restrict__ gamma,
                          const float* __restrict__ beta,
                          const float* __restrict__ confw,
                          const float* __restrict__ confb,
                          float* __restrict__ tok,
                          float* __restrict__ xout,
                          float* __restrict__ confout,
                          float* __restrict__ weight) {
  __shared__ float red[256];
  int row = blockIdx.x;
  int c = threadIdx.x;
  float inv = 1.0f / (float)(B_ * N_);
  float mu = stats[c] * inv;
  float var = stats[CO + c] * inv - mu * mu;
  float rstd = rsqrtf(var + BNEPS);
  size_t o = (size_t)row * CO + c;
  float v = (tok[o] - mu) * rstd * gamma[c] + beta[c];
  tok[o] = v;
  xout[o] = fmaxf(v, 0.f);
  red[c] = v * confw[c];
  __syncthreads();
  for (int s = 128; s > 0; s >>= 1) {
    if (c < s) red[c] += red[c + s];
    __syncthreads();
  }
  if (c == 0) {
    float cf = red[0] + confb[0];
    confout[row] = cf;
    weight[row] = expf(cf);
  }
}

// -------------------------- grid clustering --------------------------------

__global__ void k_cluster_scatter(const float* __restrict__ tok,
                                  const float* __restrict__ loc,
                                  const int* __restrict__ idx_agg,
                                  const float* __restrict__ weight,
                                  float* __restrict__ clunum,
                                  float* __restrict__ cluden,
                                  float* __restrict__ idxout) {
  int pt = blockIdx.x * 8 + (threadIdx.x >> 5);
  int lane = threadIdx.x & 31;
  int b = pt / N0, n0 = pt % N0;
  float lx = loc[(size_t)(b * N0 + n0) * 2 + 0];
  float ly = loc[(size_t)(b * N0 + n0) * 2 + 1];
  int cell = grid_cell(lx, ly, HC, WC);
  int tk = idx_agg[b * N0 + n0];
  float wpt = weight[b * N_ + tk];
  int c0 = lane * 8;
  const float* src = tok + (size_t)(b * N_ + tk) * CO + c0;
  float* dst = clunum + (size_t)(b * NS + cell) * CO + c0;
#pragma unroll
  for (int j = 0; j < 8; ++j) atomicAdd(dst + j, src[j] * wpt);
  if (lane == 0) {
    atomicAdd(cluden + (size_t)b * NS + cell, wpt);
    idxout[b * N0 + n0] = (float)cell;
  }
}

__global__ void k_xdown(const float* __restrict__ clunum,
                        const float* __restrict__ cluden,
                        float* __restrict__ out) {
  size_t i = (size_t)blockIdx.x * 256 + threadIdx.x;
  if (i >= (size_t)B_ * NS * CO) return;
  size_t cell = i / CO;
  float v = clunum[i] / (cluden[cell] + EPSF);
  out[i] = fmaxf(v, 0.f);
}

__global__ void k_awd(const float* __restrict__ loc,
                      const int* __restrict__ idx_agg,
                      const float* __restrict__ weight,
                      const float* __restrict__ cluden,
                      const float* __restrict__ aggw,
                      float* __restrict__ awd,
                      unsigned int* __restrict__ maxb) {
  int i = blockIdx.x * 256 + threadIdx.x;
  if (i >= B_ * N0) return;
  int b = i / N0;
  float lx = loc[(size_t)i * 2 + 0];
  float ly = loc[(size_t)i * 2 + 1];
  int cell = grid_cell(lx, ly, HC, WC);
  int tk = idx_agg[i];
  float wpt = weight[b * N_ + tk];
  float v = aggw[i] * wpt / (cluden[(size_t)b * NS + cell] + EPSF);
  awd[i] = v;
  atomicMax(maxb + b, __float_as_uint(v));  // all values >= 0
}

__global__ void k_awd_norm(const float* __restrict__ awd,
                           const unsigned int* __restrict__ maxb,
                           float* __restrict__ out) {
  int i = blockIdx.x * 256 + threadIdx.x;
  if (i >= B_ * N0) return;
  int b = i / N0;
  out[i] = awd[i] / __uint_as_float(maxb[b]);
}

// -------------------------- host launcher ----------------------------------

extern "C" void kernel_launch(void* const* d_in, const int* in_sizes, int n_in,
                              void* d_out, int out_size, void* d_ws, size_t ws_size,
                              hipStream_t stream) {
  (void)in_sizes; (void)n_in; (void)out_size; (void)ws_size;
  const float* x = (const float*)d_in[0];
  const float* loc = (const float*)d_in[1];
  const int* idx_agg = (const int*)d_in[2];
  const float* aggw = (const float*)d_in[3];
  const float* convw = (const float*)d_in[4];
  const float* convb = (const float*)d_in[5];
  const float* skipw = (const float*)d_in[6];
  const float* gamma = (const float*)d_in[7];
  const float* beta = (const float*)d_in[8];
  const float* confw = (const float*)d_in[9];
  const float* confb = (const float*)d_in[10];

  char* ws = (char*)d_ws;
  size_t off = 0;
  auto alloc = [&](size_t bytes) -> char* {
    char* p = ws + off;
    off += (bytes + 255) & ~(size_t)255;
    return p;
  };

  float* cnt = (float*)alloc((size_t)B_ * H_ * W_ * 4);                // 1 MB
  float* ssum = (float*)alloc((size_t)B_ * H_ * W_ * CI * 4);          // 128 MB
  unsigned short* xmapb = (unsigned short*)alloc((size_t)B_ * H_ * W_ * CI * 2); // 64 MB
  float* y = (float*)alloc((size_t)B_ * HD * WD * CO * 4);             // 64 MB
  float* tok = (float*)alloc((size_t)B_ * N_ * CO * 4);                // 64 MB
  float* tokden = (float*)alloc((size_t)B_ * N_ * 4);
  float* stats = (float*)alloc(2 * CO * 4);
  float* clunum = (float*)alloc((size_t)B_ * NS * CO * 4);             // 16 MB
  float* cluden = (float*)alloc((size_t)B_ * NS * 4);
  unsigned short* wq = (unsigned short*)alloc((size_t)CO * 9 * CI * 2);
  unsigned short* sw = (unsigned short*)alloc((size_t)CO * CI * 2);
  unsigned short* xb = (unsigned short*)alloc((size_t)B_ * N_ * CI * 2); // 16 MB
  float* weight = (float*)alloc((size_t)B_ * N_ * 4);
  float* awd = (float*)alloc((size_t)B_ * N0 * 4);
  unsigned int* maxb = (unsigned int*)alloc(B_ * 4);

  float* out = (float*)d_out;
  float* o_xdown = out;                                 // [4,4096,256]
  float* o_xout = o_xdown + (size_t)B_ * NS * CO;       // [4,16384,256]
  float* o_conf = o_xout + (size_t)B_ * N_ * CO;        // [4,16384,1]
  float* o_awd = o_conf + (size_t)B_ * N_;              // [4,65536,1]
  float* o_idx = o_awd + (size_t)B_ * N0;               // [4,65536]

  // zero accumulators
  hipMemsetAsync(cnt, 0, (size_t)B_ * H_ * W_ * 4, stream);
  hipMemsetAsync(ssum, 0, (size_t)B_ * H_ * W_ * CI * 4, stream);
  hipMemsetAsync(tok, 0, (size_t)B_ * N_ * CO * 4, stream);
  hipMemsetAsync(tokden, 0, (size_t)B_ * N_ * 4, stream);
  hipMemsetAsync(stats, 0, 2 * CO * 4, stream);
  hipMemsetAsync(clunum, 0, (size_t)B_ * NS * CO * 4, stream);
  hipMemsetAsync(cluden, 0, (size_t)B_ * NS * 4, stream);
  hipMemsetAsync(maxb, 0, B_ * 4, stream);

  // pack bf16 operands
  k_pack_convw<<<(CO * 9 * CI + 255) / 256, 256, 0, stream>>>(convw, wq);
  k_pack_skipw<<<(CO * CI + 255) / 256, 256, 0, stream>>>(skipw, sw);
  k_pack_x<<<(B_ * N_ * CI + 255) / 256, 256, 0, stream>>>(x, xb);

  // token2map
  k_token2map_scatter<<<B_ * N0 / 8, 256, 0, stream>>>(x, loc, idx_agg, ssum, cnt);
  k_map_norm<<<(int)(((size_t)B_ * H_ * W_ * CI + 255) / 256), 256, 0, stream>>>(ssum, cnt, xmapb);

  // 3x3 s2 conv via WMMA implicit GEMM (block: 32 pixels x full CO)
  k_conv_wmma<<<B_ * HD * (WD / 32), 256, 0, stream>>>(xmapb, wq, convb, y);

  // map2token + skip linear
  k_map2token_scatter<<<B_ * N0 / 8, 256, 0, stream>>>(y, loc, idx_agg, aggw, tok, tokden);
  k_skip_wmma<<<B_ * N_ / 16, 256, 0, stream>>>(xb, sw, tokden, tok);

  // BN + conf
  k_bn_stats<<<B_ * N_ / 256, 256, 0, stream>>>(tok, stats);
  k_bn_conf<<<B_ * N_, 256, 0, stream>>>(stats, gamma, beta, confw, confb,
                                         tok, o_xout, o_conf, weight);

  // grid clustering
  k_cluster_scatter<<<B_ * N0 / 8, 256, 0, stream>>>(tok, loc, idx_agg, weight,
                                                     clunum, cluden, o_idx);
  k_xdown<<<(int)(((size_t)B_ * NS * CO + 255) / 256), 256, 0, stream>>>(clunum, cluden, o_xdown);
  k_awd<<<(B_ * N0 + 255) / 256, 256, 0, stream>>>(loc, idx_agg, weight, cluden, aggw, awd, maxb);
  k_awd_norm<<<(B_ * N0 + 255) / 256, 256, 0, stream>>>(awd, maxb, o_awd);
}
